// CondensationLossTiger_59107339927841
// MI455X (gfx1250) — compile-verified, matrix operand order checked by reading, not run on hardware
//
#include <hip/hip_runtime.h>
#include <math.h>

#define NH 50000
#define NP 1500
#define KP 1504              // 94 * 16 (padded pid columns)
#define NTI (NH / 16)        // 3125 row tiles (exact)
#define NTJ (KP / 16)        // 94 col tiles
#define Q_MIN   0.01f
#define PT_THLD 0.9f
#define MAX_ETA 4.0f
#define EPSF    1e-9f

typedef __attribute__((ext_vector_type(2)))  float    v2f;
typedef __attribute__((ext_vector_type(8)))  float    v8f;
typedef __attribute__((ext_vector_type(16))) _Float16 v16h;

#if __has_builtin(__builtin_amdgcn_wmma_f32_16x16x4_f32)
#define USE_WMMA_F32 1
#else
#define USE_WMMA_F32 0
#endif

__device__ __forceinline__ float fast_sqrtf(float v) {
#if __has_builtin(__builtin_amdgcn_sqrtf)
  return __builtin_amdgcn_sqrtf(v);   // raw v_sqrt_f32, <=1 ulp
#else
  return sqrtf(v);
#endif
}

// ---------------------------------------------------------------- init
__global__ void k_init(unsigned long long* argkey, unsigned* attc, unsigned* pres,
                       unsigned* su, float* sf) {
  int k = blockIdx.x * blockDim.x + threadIdx.x;
  if (k < KP) { argkey[k] = 0ull; attc[k] = 0u; pres[k] = 0u; }
  if (k < 8)  { su[k] = 0u; sf[k] = 0.0f; }
}

// ---------------------------------------------------------------- per-hit
// rowdat[i] = { q_i, ||x_i||^2, bitcast(pid_i), 0 }  -> one b128 load in hot loop
__global__ void k_hits(const float* __restrict__ beta, const float* __restrict__ x,
                       const int* __restrict__ pid, const int* __restrict__ recon,
                       const float* __restrict__ pt, const float* __restrict__ eta,
                       float4* __restrict__ rowdat,
                       unsigned long long* argkey, unsigned* attc, unsigned* pres,
                       unsigned* su, float* sf) {
  int i = blockIdx.x * blockDim.x + threadIdx.x;
  if (i >= NH) return;
  float b = beta[i];
  float t = atanhf(b);
  float qi = t * t + Q_MIN;
  float s = 0.f;
  #pragma unroll
  for (int d = 0; d < 8; ++d) { float v = x[(size_t)i * 8 + d]; s += v * v; }
  int p = pid[i];
  float4 rd; rd.x = qi; rd.y = s; rd.z = __int_as_float(p); rd.w = 0.f;
  rowdat[i] = rd;
  bool mk = (pt[i] > PT_THLD) && (p > 0) && (recon[i] > 0) && (fabsf(eta[i]) < MAX_ETA);
  if (p >= 1 && p <= NP) {
    int k = p - 1;
    atomicAdd(&attc[k], 1u);
    if (mk) atomicOr(&pres[k], 1u);
    // positive-float bits are order-preserving as uint; ~i so ties pick smallest i
    unsigned long long key = ((unsigned long long)__float_as_uint(qi) << 32)
                           | (unsigned long long)(0xFFFFFFFFu - (unsigned)i);
    atomicMax(&argkey[k], key);
  }
  if (p <= 0) {
    atomicAdd(&sf[1], b);   // noise beta sum
    atomicAdd(&su[1], 1u);  // noise count
  }
}

// ---------------------------------------------------------------- per-pid pass A
__global__ void k_pidA(const unsigned long long* __restrict__ argkey,
                       const unsigned* __restrict__ pres, const float* __restrict__ beta,
                       unsigned* alpha, unsigned* su, float* sf) {
  int k = blockIdx.x * blockDim.x + threadIdx.x;
  if (k >= NP) return;
  unsigned long long key = argkey[k];
  unsigned a = key ? (0xFFFFFFFFu - (unsigned)(key & 0xFFFFFFFFull)) : 0u; // absent -> 0 (argmax of zeros)
  alpha[k] = a;
  if (pres[k]) {
    atomicAdd(&su[0], 1u);               // n_unique
    atomicAdd(&sf[0], 1.0f - beta[a]);   // l_coward numerator
  }
}

// ---------------------------------------------------------------- per-pid pass B
// coldat[k] = { ||x_k||^2, q_k, inv_att_norm_k (incl colw/n_unique), inv_rep_norm_k }
__global__ void k_pidB(const unsigned* __restrict__ alpha, const unsigned* __restrict__ pres,
                       const unsigned* __restrict__ attc, const float* __restrict__ x,
                       const float4* __restrict__ rowdat, const unsigned* __restrict__ su,
                       float* __restrict__ xk, float4* __restrict__ coldat) {
  int k = blockIdx.x * blockDim.x + threadIdx.x;
  if (k >= KP) return;
  float4 cd; cd.x = 0.f; cd.y = 0.f; cd.z = 0.f; cd.w = 0.f;
  if (k < NP) {
    unsigned a = alpha[k];
    float s = 0.f;
    #pragma unroll
    for (int d = 0; d < 8; ++d) { float v = x[(size_t)a * 8 + d]; xk[(size_t)k * 8 + d] = v; s += v * v; }
    cd.x = s;
    cd.y = rowdat[a].x;                       // q_k
    float nu  = (float)su[0];
    float pf  = pres[k] ? 1.0f : 0.0f;        // colw folded into the norms
    float cnt = (float)attc[k];
    cd.z = pf / ((cnt + EPSF) * nu);
    cd.w = pf / (((float)NH - cnt + EPSF) * nu);
  } else {
    #pragma unroll
    for (int d = 0; d < 8; ++d) xk[(size_t)k * 8 + d] = 0.f;  // pres[k] already 0
  }
  coldat[k] = cd;
}

// ---------------------------------------------------------------- pair kernel (WMMA)
__global__ void __launch_bounds__(256) k_pairs(
    const float*  __restrict__ x,   const float4* __restrict__ rowdat,
    const float*  __restrict__ xk,  const float4* __restrict__ coldat,
    const unsigned* __restrict__ pres,
    unsigned* su, float* sf) {
  const int tid  = threadIdx.x;
  const int lane = tid & 31;
  const int half = lane >> 4;   // half-wave split (K for A/B, M-block for D per ISA layout)
  const int lp   = lane & 15;
  const int wid  = blockIdx.x * (blockDim.x >> 5) + (tid >> 5);
  const int nw   = gridDim.x * (blockDim.x >> 5);
  const int tj   = blockIdx.y;
  const int n    = tj * 16 + lp;              // this lane's pid column (D-matrix N = lane%16)

  // column-constant data, hoisted for the whole ti sweep
  const float4 cd = coldat[n];
  const float nk_n = cd.x, qk_n = cd.y, ia_n = cd.z, ir_n = cd.w;
  const unsigned pr_n = pres[n];
  const int pid_n = n + 1;
  const float* br = xk + (size_t)n * 8;
#if USE_WMMA_F32
  // B 4x16 f32: VGPR0 = K{0|2}, VGPR1 = K{1|3} split by half; lane%16 = N
  v2f b0, b1;
  b0.x = br[2 * half];     b0.y = br[2 * half + 1];
  b1.x = br[4 + 2 * half]; b1.y = br[4 + 2 * half + 1];
#else
  v16h bh;
  #pragma unroll
  for (int e = 0; e < 16; ++e) bh[e] = (_Float16)0.f;
  if (half == 0) {
    #pragma unroll
    for (int d = 0; d < 8; ++d) bh[d] = (_Float16)br[d];
  }
#endif

  float vatt = 0.f, vrep = 0.f;
  unsigned nrep = 0u;

  for (int ti = wid; ti < NTI; ti += nw) {   // uniform per-wave bounds -> EXEC all-ones at WMMA
    const int i0 = ti * 16;
    const float* ar = x + (size_t)(i0 + lp) * 8;
    v8f acc = {};
#if USE_WMMA_F32
    v2f a0, a1;
    a0.x = ar[2 * half];     a0.y = ar[2 * half + 1];
    a1.x = ar[4 + 2 * half]; a1.y = ar[4 + 2 * half + 1];
    acc = __builtin_amdgcn_wmma_f32_16x16x4_f32(false, a0, false, b0, (short)0, acc, false, false);
    acc = __builtin_amdgcn_wmma_f32_16x16x4_f32(false, a1, false, b1, (short)0, acc, false, false);
#else
    v16h ah;
    #pragma unroll
    for (int e = 0; e < 16; ++e) ah[e] = (_Float16)0.f;
    if (half == 0) {
      #pragma unroll
      for (int d = 0; d < 8; ++d) ah[d] = (_Float16)ar[d];
    }
    acc = __builtin_amdgcn_wmma_f32_16x16x32_f16(false, ah, false, bh, (short)0, acc, false, false);
#endif
    // D layout: VGPR r holds M = r + 8*half, N = lane%16.  Branchless epilogue.
    #pragma unroll
    for (int r = 0; r < 8; ++r) {
      const float4 rd = rowdat[i0 + r + 8 * half];   // {q_m, nx_m, pid_m}
      const int   pm  = __float_as_int(rd.z);
      float d2 = fmaxf(fmaf(-2.0f, acc[r], rd.y + nk_n), 0.0f);
      const float dist = fast_sqrtf(d2 + 1e-12f);
      const float qw   = rd.x * qk_n;
      const bool  att  = (pm == pid_n);
      const bool  rep  = (!att) && (dist < 1.0f);
      vatt = fmaf(att ? qw * d2 : 0.0f, ia_n, vatt);
      vrep = fmaf(rep ? qw * (1.0f - dist) : 0.0f, ir_n, vrep);
      nrep += rep ? pr_n : 0u;
    }
  }

  __shared__ float    sva[256];
  __shared__ float    svr[256];
  __shared__ unsigned snr[256];
  sva[tid] = vatt; svr[tid] = vrep; snr[tid] = nrep;
  __syncthreads();
  for (int s = 128; s > 0; s >>= 1) {
    if (tid < s) { sva[tid] += sva[tid + s]; svr[tid] += svr[tid + s]; snr[tid] += snr[tid + s]; }
    __syncthreads();
  }
  if (tid == 0) {
    atomicAdd(&sf[2], sva[0]);
    atomicAdd(&sf[3], svr[0]);
    atomicAdd(&su[2], snr[0]);
  }
}

// ---------------------------------------------------------------- finalize
__global__ void k_final(const unsigned* su, const float* sf, float* out) {
  if (threadIdx.x == 0 && blockIdx.x == 0) {
    out[0] = sf[2];                              // v_att
    out[1] = sf[3];                              // v_rep
    out[2] = sf[0] / (float)su[0];               // l_coward
    unsigned nc = su[1]; if (nc == 0u) nc = 1u;
    out[3] = sf[1] / (float)nc;                  // l_noise
    out[4] = (float)su[2];                       // n_rep
  }
}

// ---------------------------------------------------------------- launch
extern "C" void kernel_launch(void* const* d_in, const int* in_sizes, int n_in,
                              void* d_out, int out_size, void* d_ws, size_t ws_size,
                              hipStream_t stream) {
  (void)in_sizes; (void)n_in; (void)out_size; (void)ws_size;
  const float* beta  = (const float*)d_in[0];
  const float* x     = (const float*)d_in[1];
  const int*   pid   = (const int*)d_in[2];
  const int*   recon = (const int*)d_in[3];
  const float* pt    = (const float*)d_in[4];
  const float* eta   = (const float*)d_in[5];
  float* out = (float*)d_out;

  char* w = (char*)d_ws;
  size_t off = 0;
  auto take = [&](size_t bytes, size_t align) -> char* {
    off = (off + align - 1) & ~(align - 1);
    char* p = w + off; off += bytes; return p;
  };
  unsigned long long* argkey = (unsigned long long*)take((size_t)KP * 8, 8);
  float4*   rowdat = (float4*)take((size_t)NH * 16, 16);
  float4*   coldat = (float4*)take((size_t)KP * 16, 16);
  float*    xk     = (float*)take((size_t)KP * 8 * 4, 16);
  unsigned* attc   = (unsigned*)take((size_t)KP * 4, 4);
  unsigned* pres   = (unsigned*)take((size_t)KP * 4, 4);
  unsigned* alpha  = (unsigned*)take((size_t)KP * 4, 4);
  unsigned* su     = (unsigned*)take(8 * 4, 4);   // [0]=n_unique [1]=noise_cnt [2]=n_rep
  float*    sf     = (float*)take(8 * 4, 4);      // [0]=lcow_num [1]=noise_sum [2]=v_att [3]=v_rep

  k_init <<<(KP + 255) / 256, 256, 0, stream>>>(argkey, attc, pres, su, sf);
  k_hits <<<(NH + 255) / 256, 256, 0, stream>>>(beta, x, pid, recon, pt, eta,
                                                rowdat, argkey, attc, pres, su, sf);
  k_pidA <<<(NP + 255) / 256, 256, 0, stream>>>(argkey, pres, beta, alpha, su, sf);
  k_pidB <<<(KP + 255) / 256, 256, 0, stream>>>(alpha, pres, attc, x, rowdat, su,
                                                xk, coldat);
  dim3 g(40, NTJ, 1);                            // 320 waves per column tile x 94 tiles
  k_pairs<<<g, 256, 0, stream>>>(x, rowdat, xk, coldat, pres, su, sf);
  k_final<<<1, 1, 0, stream>>>(su, sf, out);
}
